// MultiHeadAttention_44573170598915
// MI455X (gfx1250) — matrix operand order repaired and checked by reading, compiled, not run on hardware
//
#include <hip/hip_runtime.h>
#include <hip/hip_fp16.h>

typedef __attribute__((ext_vector_type(16))) _Float16 v16h;
typedef __attribute__((ext_vector_type(8)))  _Float16 v8h;
typedef __attribute__((ext_vector_type(8)))  float    v8f;
typedef __attribute__((ext_vector_type(4)))  float    v4f;

#define WMMA_F16F32(A, B, C) \
    __builtin_amdgcn_wmma_f32_16x16x32_f16(false, (A), false, (B), (short)0, (C), false, false)

// ---------------------------------------------------------------------------
// Operand loaders (CDNA5 wave32 WMMA register layouts, CDNA5 ISA)
//
// A-matrix 16x32 f16:  lane L holds row m = L&15; element i holds
//   K = i + (i&8) + 8*(L>=16)   (two contiguous runs of 8)
// B-matrix 32x16 f16:  lane L holds col n = L&15; element i holds
//   K = i + (i&8) + 8*(L>=16)   (two contiguous runs of 8)
// C/D 16x16 f32:       lane L, vgpr r -> m = r + 8*(L>=16), n = L&15
// ---------------------------------------------------------------------------

// A tile from f32 row-major source (src = top-left of 16x32 tile, leading dim ld)
__device__ __forceinline__ v16h load_tileA_f32(const float* __restrict__ src, long ld, int lane) {
    const int c0 = (lane & 16) >> 1;            // 0 or 8
    const float* p = src + (long)(lane & 15) * ld;
    v4f a0 = *(const v4f*)(p + c0);
    v4f a1 = *(const v4f*)(p + c0 + 4);
    v4f b0 = *(const v4f*)(p + 16 + c0);
    v4f b1 = *(const v4f*)(p + 16 + c0 + 4);
    v16h r;
#pragma unroll
    for (int j = 0; j < 4; ++j) {
        r[j]      = (_Float16)a0[j];
        r[4 + j]  = (_Float16)a1[j];
        r[8 + j]  = (_Float16)b0[j];
        r[12 + j] = (_Float16)b1[j];
    }
    return r;
}

// A tile from f16 row-major source
__device__ __forceinline__ v16h load_tileA_f16(const _Float16* __restrict__ src, long ld, int lane) {
    const int c0 = (lane & 16) >> 1;
    const _Float16* p = src + (long)(lane & 15) * ld;
    v8h a = *(const v8h*)(p + c0);
    v8h b = *(const v8h*)(p + 16 + c0);
    v16h r;
#pragma unroll
    for (int j = 0; j < 8; ++j) { r[j] = a[j]; r[8 + j] = b[j]; }
    return r;
}

// B tile from f16 "n-major" source: row index = n (0..15), columns = K.
// src points at (n=0, k=0) of the 32x16 logical tile stored as [n][k].
__device__ __forceinline__ v16h load_tileB_f16(const _Float16* __restrict__ src, long ld, int lane) {
    const _Float16* p = src + (long)(lane & 15) * ld + ((lane >> 4) << 3);
    v8h a = *(const v8h*)(p);
    v8h b = *(const v8h*)(p + 16);
    v16h r;
#pragma unroll
    for (int j = 0; j < 8; ++j) { r[j] = a[j]; r[8 + j] = b[j]; }
    return r;
}

// ---------------------------------------------------------------------------
// Kernel 1: weight prep.
//  WqT/WkT/WvT : [64][512] f16,  WT[j][c] = W[c][j]       (B-operand friendly)
//  WoT         : [512][64] f16,  WoT[j][c] = sum_h Wo[h*64+c][j]  (folded heads)
// ---------------------------------------------------------------------------
__global__ void prep_weights_kernel(const float* __restrict__ Wq, const float* __restrict__ Wk,
                                    const float* __restrict__ Wv, const float* __restrict__ Wo,
                                    _Float16* __restrict__ WqT, _Float16* __restrict__ WkT,
                                    _Float16* __restrict__ WvT, _Float16* __restrict__ WoT) {
    int t = blockIdx.x * blockDim.x + threadIdx.x;
    if (t >= 64 * 512) return;
    int c = t % 512, j = t / 512;               // j < 64
    WqT[j * 512 + c] = (_Float16)Wq[c * 64 + j];
    WkT[j * 512 + c] = (_Float16)Wk[c * 64 + j];
    WvT[j * 512 + c] = (_Float16)Wv[c * 64 + j];
    int cc = t % 64, jj = t / 64;               // jj < 512
    float s = 0.f;
#pragma unroll
    for (int h = 0; h < 8; ++h) s += Wo[(h * 64 + cc) * 512 + jj];
    WoT[jj * 64 + cc] = (_Float16)s;
}

// ---------------------------------------------------------------------------
// Kernel 2: projections. One wave per 16 rows; one fully-inlined straight-line
// instance per matrix (no runtime pointer selection -> global_load + no branches).
//   TRANSPOSED=false : dst row-major [row][64] f16      (qp, kp)
//   TRANSPOSED=true  : dst per-batch transposed [b][64][n] f16 (vpT)
// ---------------------------------------------------------------------------
template <bool TRANSPOSED>
__device__ __forceinline__ void proj_one(const float* __restrict__ src,
                                         const _Float16* __restrict__ wt,
                                         const float* __restrict__ bias,
                                         _Float16* __restrict__ dst,
                                         long row0, long batch, long rin, int n, int lane) {
    const int m = lane & 15, half = lane >> 4;
    v8f acc[4] = {};
    for (int kc = 0; kc < 16; ++kc) {                           // K = 512, chunks of 32
        v16h a = load_tileA_f32(src + row0 * 512 + kc * 32, 512, lane);
#pragma unroll
        for (int nt = 0; nt < 4; ++nt) {                        // 64 output cols
            v16h bm = load_tileB_f16(wt + (long)(nt * 16) * 512 + kc * 32, 512, lane);
            acc[nt] = WMMA_F16F32(a, bm, acc[nt]);
        }
    }
#pragma unroll
    for (int nt = 0; nt < 4; ++nt) {
        const float bb = bias[nt * 16 + m];
#pragma unroll
        for (int r = 0; r < 8; ++r) {
            const float val = acc[nt][r] + bb;
            const long  rr  = r + 8 * half;                     // D-layout row
            if (TRANSPOSED)
                dst[batch * 64 * n + (long)(nt * 16 + m) * n + rin + rr] = (_Float16)val;
            else
                dst[(row0 + rr) * 64 + nt * 16 + m] = (_Float16)val;
        }
    }
}

__global__ __launch_bounds__(32) void proj_kernel(
    const float* __restrict__ q, const float* __restrict__ k, const float* __restrict__ v,
    const _Float16* __restrict__ WqT, const _Float16* __restrict__ WkT,
    const _Float16* __restrict__ WvT,
    const float* __restrict__ bq, const float* __restrict__ bk, const float* __restrict__ bv,
    _Float16* __restrict__ qp, _Float16* __restrict__ kp, _Float16* __restrict__ vpT, int n) {
    const int lane = threadIdx.x;
    const long row0 = (long)blockIdx.x * 16;
    const long batch = row0 / n, rin = row0 % n;
    proj_one<false>(q, WqT, bq, qp,  row0, batch, rin, n, lane);
    proj_one<false>(k, WkT, bk, kp,  row0, batch, rin, n, lane);
    proj_one<true >(v, WvT, bv, vpT, row0, batch, rin, n, lane);
}

// ---------------------------------------------------------------------------
// Kernel 3: flash attention. One wave handles a 16-query tile, streaming all
// n keys in chunks of 32 with online softmax. Scores never touch HBM.
// ---------------------------------------------------------------------------
__global__ __launch_bounds__(32) void attn_kernel(
    const _Float16* __restrict__ qp, const _Float16* __restrict__ kp,
    const _Float16* __restrict__ vpT, _Float16* __restrict__ o, int n) {
    __shared__ float sS[16 * 32];
    __shared__ float sM[16], sL[16], sF[16];

    const int lane = threadIdx.x;
    const int m = lane & 15, half = lane >> 4, c0 = (lane & 16) >> 1;
    const long row0  = (long)blockIdx.x * 16;
    const long batch = row0 / n;
    const float scale = 0.125f;                    // 1/sqrt(64)

    // Q tile in A-layout registers, held for the whole kernel (K = 64 -> 2 chunks)
    v16h qa0 = load_tileA_f16(qp + row0 * 64, 64, lane);
    v16h qa1 = load_tileA_f16(qp + row0 * 64 + 32, 64, lane);

    const _Float16* kb = kp + batch * (long)n * 64;   // [key][64]
    const _Float16* vb = vpT + batch * 64 * (long)n;  // [d][n]

    v8f acc[4] = {};                                  // 16 x 64 output accumulator
    if (lane < 16) { sM[lane] = -1e30f; sL[lane] = 0.f; }
    __syncthreads();

    for (long kc = 0; kc < n; kc += 32) {
        // S = Q * K^T for 32 keys (2 N-tiles, K-loop over d = 64)
#pragma unroll
        for (int nt = 0; nt < 2; ++nt) {
            v16h b0 = load_tileB_f16(kb + (kc + nt * 16) * 64, 64, lane);
            v16h b1 = load_tileB_f16(kb + (kc + nt * 16) * 64 + 32, 64, lane);
            v8f s = {};
            s = WMMA_F16F32(qa0, b0, s);
            s = WMMA_F16F32(qa1, b1, s);
#pragma unroll
            for (int r = 0; r < 8; ++r)
                sS[(r + 8 * half) * 32 + nt * 16 + m] = s[r] * scale;
        }
        __syncthreads();

        // online softmax: lanes 0..15 own one query row each
        if (lane < 16) {
            float mx = sM[lane];
#pragma unroll
            for (int j = 0; j < 32; ++j) mx = fmaxf(mx, sS[lane * 32 + j]);
            const float f = __expf(sM[lane] - mx);
            float sum = sL[lane] * f;
#pragma unroll
            for (int j = 0; j < 32; ++j) {
                const float p = __expf(sS[lane * 32 + j] - mx);
                sS[lane * 32 + j] = p;
                sum += p;
            }
            sM[lane] = mx; sL[lane] = sum; sF[lane] = f;
        }
        __syncthreads();

        // rescale running accumulator by per-row factor
#pragma unroll
        for (int r = 0; r < 8; ++r) {
            const float f = sF[r + 8 * half];
#pragma unroll
            for (int nt = 0; nt < 4; ++nt) acc[nt][r] *= f;
        }

        // P tile (A-layout) from LDS
        v16h pa;
#pragma unroll
        for (int j = 0; j < 8; ++j) {
            pa[j]     = (_Float16)sS[m * 32 + c0 + j];
            pa[8 + j] = (_Float16)sS[m * 32 + 16 + c0 + j];
        }

        // O += P * V  (V transposed -> contiguous B loads), 4 d-tiles
#pragma unroll
        for (int nt = 0; nt < 4; ++nt) {
            v16h vv = load_tileB_f16(vb + (long)(nt * 16) * n + kc, n, lane);
            acc[nt] = WMMA_F16F32(pa, vv, acc[nt]);
        }
        __syncthreads();
    }

    // normalize and emit o row-major f16 [row][64]
#pragma unroll
    for (int r = 0; r < 8; ++r) {
        const float invl = 1.0f / sL[r + 8 * half];
#pragma unroll
        for (int nt = 0; nt < 4; ++nt)
            o[(row0 + r + 8 * half) * 64 + nt * 16 + m] = (_Float16)(acc[nt][r] * invl);
    }
}

// ---------------------------------------------------------------------------
// Kernel 4: out = o @ Wo_eff + bo   ((rows x 64) * (64 x 512), f32 out)
// ---------------------------------------------------------------------------
__global__ __launch_bounds__(32) void out_kernel(
    const _Float16* __restrict__ o, const _Float16* __restrict__ WoT,
    const float* __restrict__ bo, float* __restrict__ out) {
    const int lane = threadIdx.x;
    const int m = lane & 15, half = lane >> 4;
    const long row0 = (long)blockIdx.x * 16;

    v16h a0 = load_tileA_f16(o + row0 * 64, 64, lane);
    v16h a1 = load_tileA_f16(o + row0 * 64 + 32, 64, lane);

    for (int nt = 0; nt < 32; ++nt) {                 // 512 output cols
        v16h b0 = load_tileB_f16(WoT + (long)(nt * 16) * 64, 64, lane);
        v16h b1 = load_tileB_f16(WoT + (long)(nt * 16) * 64 + 32, 64, lane);
        v8f acc = {};
        acc = WMMA_F16F32(a0, b0, acc);
        acc = WMMA_F16F32(a1, b1, acc);
        const float bb = bo[nt * 16 + m];
#pragma unroll
        for (int r = 0; r < 8; ++r)
            out[(row0 + r + 8 * half) * 512 + nt * 16 + m] = acc[r] + bb;
    }
}

// ---------------------------------------------------------------------------
extern "C" void kernel_launch(void* const* d_in, const int* in_sizes, int n_in,
                              void* d_out, int out_size, void* d_ws, size_t ws_size,
                              hipStream_t stream) {
    (void)n_in; (void)out_size; (void)ws_size;
    const float* q  = (const float*)d_in[0];
    const float* k  = (const float*)d_in[1];
    const float* v  = (const float*)d_in[2];
    const float* Wq = (const float*)d_in[3];
    const float* bq = (const float*)d_in[4];
    const float* Wk = (const float*)d_in[5];
    const float* bk = (const float*)d_in[6];
    const float* Wv = (const float*)d_in[7];
    const float* bv = (const float*)d_in[8];
    const float* Wo = (const float*)d_in[9];
    const float* bo = (const float*)d_in[10];
    float* out = (float*)d_out;

    const long rows = (long)in_sizes[0] / 512;   // b*n = 8192
    const int  n    = 2048;                      // sequence length per batch

    // workspace carve-up (256B aligned)
    char* ws = (char*)d_ws;
    const size_t szAct = (size_t)rows * 64 * sizeof(_Float16);   // 1 MB each
    _Float16* qp  = (_Float16*)(ws);                 ws += (szAct + 255) & ~255ull;
    _Float16* kp  = (_Float16*)(ws);                 ws += (szAct + 255) & ~255ull;
    _Float16* vpT = (_Float16*)(ws);                 ws += (szAct + 255) & ~255ull;
    _Float16* o   = (_Float16*)(ws);                 ws += (szAct + 255) & ~255ull;
    const size_t szW = 64 * 512 * sizeof(_Float16);  // 64 KB each
    _Float16* WqT = (_Float16*)(ws);                 ws += szW;
    _Float16* WkT = (_Float16*)(ws);                 ws += szW;
    _Float16* WvT = (_Float16*)(ws);                 ws += szW;
    _Float16* WoT = (_Float16*)(ws);                 ws += szW;

    prep_weights_kernel<<<(64 * 512 + 255) / 256, 256, 0, stream>>>(
        Wq, Wk, Wv, Wo, WqT, WkT, WvT, WoT);
    proj_kernel<<<(int)(rows / 16), 32, 0, stream>>>(
        q, k, v, WqT, WkT, WvT, bq, bk, bv, qp, kp, vpT, n);
    attn_kernel<<<(int)(rows / 16), 32, 0, stream>>>(qp, kp, vpT, o, n);
    out_kernel<<<(int)(rows / 16), 32, 0, stream>>>(o, WoT, bo, out);
}
